// CouplingLayer_53360673686024
// MI455X (gfx1250) — matrix operand order repaired and checked by reading, not compiled, not run on hardware
//
#include <hip/hip_runtime.h>
#include <hip/hip_bf16.h>
#include <math.h>

// ---------------------------------------------------------------------------
// Model dims (compile-time constants from the reference)
// ---------------------------------------------------------------------------
#define NL 4
#define DMODEL 256
#define DI 512
#define DS 16
#define DR 16
#define KC 4
#define BQ 2
#define LSEQ 1024
#define NTOK (BQ * LSEQ)   // 2048 rows

typedef __attribute__((ext_vector_type(16))) __bf16 v16bf;
typedef __attribute__((ext_vector_type(8)))  float  v8f;

// Pack 8 consecutive f32 (two float4s) into bf16 fragment slots [o, o+8).
__device__ __forceinline__ void pack8(v16bf& f, int o, float4 x, float4 y)
{
    f[o + 0] = (__bf16)x.x; f[o + 1] = (__bf16)x.y;
    f[o + 2] = (__bf16)x.z; f[o + 3] = (__bf16)x.w;
    f[o + 4] = (__bf16)y.x; f[o + 5] = (__bf16)y.y;
    f[o + 6] = (__bf16)y.z; f[o + 7] = (__bf16)y.w;
}

// ---------------------------------------------------------------------------
// WMMA GEMM:  C[m][n] = act( sum_k A[m][k] * W[n][k] + bias[n] )
// Requirements: K % 32 == 0, M % 16 == 0, (N/16) % NT == 0, rows 16B-aligned.
// One wave computes a 16 x (16*NT) strip: the A fragment is loaded once per
// K-step and reused across NT v_wmma_f32_16x16x32_bf16 issues.  NT is a
// template constant so the inner loop is branch-free (EXEC stays all-1s for
// every WMMA, per ISA 7.12 restrictions) and accumulators never get shuffled
// through saveexec regions.  act: 0=none, 1=bias+softplus, 2=bias.
// ---------------------------------------------------------------------------
template <int NT>
__global__ __launch_bounds__(128)
void gemm_wmma_kernel(const float* __restrict__ A, int lda,
                      const float* __restrict__ W, int ldw,
                      float* __restrict__ C, int ldc,
                      int M, int N, int K,
                      const float* __restrict__ bias, int act)
{
    const int wave    = threadIdx.x >> 5;
    const int lane    = threadIdx.x & 31;
    const int tilesN  = N >> 4;
    const int groupsN = tilesN / NT;
    const int total   = (M >> 4) * groupsN;
    const int job = blockIdx.x * (blockDim.x >> 5) + wave;
    if (job >= total) return;            // wave-uniform; grids are exact

    const int tm = job / groupsN;
    const int gn = job - tm * groupsN;
    const int m0 = tm << 4;
    const int lid = lane & 15;          // M index (A) / N index (B,C)
    const int hf  = lane >> 4;          // half-wave selector
    // A 16x32 bf16 layout (ISA 7.12.2): lanes 0-15 hold K {0..7,16..23},
    // lanes 16-31 hold K {8..15,24..31} -> two contiguous 8-float runs.
    const int baseA0 = hf ? 8 : 0;
    const int baseA1 = hf ? 24 : 16;

    v8f acc[NT];
#pragma unroll
    for (int j = 0; j < NT; ++j) acc[j] = v8f{};

    const float* arowBase = A + (size_t)(m0 + lid) * lda;
    const float* browBase = W + (size_t)((gn * NT) * 16 + lid) * ldw + hf * 8;

    for (int k0 = 0; k0 < K; k0 += 32) {
        const float* ar = arowBase + k0;
        float4 a0 = *(const float4*)(ar + baseA0);
        float4 a1 = *(const float4*)(ar + baseA0 + 4);
        float4 a2 = *(const float4*)(ar + baseA1);
        float4 a3 = *(const float4*)(ar + baseA1 + 4);
        v16bf afrag;
        pack8(afrag, 0, a0, a1);
        pack8(afrag, 8, a2, a3);

#pragma unroll
        for (int j = 0; j < NT; ++j) {
            // B 32x16 bf16 layout: lane=N; lanes 0-15 hold K {0..7,16..23},
            // lanes 16-31 hold K {8..15,24..31} -> two contiguous 8-float runs.
            const float* br = browBase + (size_t)j * 16 * ldw + k0;
            float4 b0 = *(const float4*)(br);
            float4 b1 = *(const float4*)(br + 4);
            float4 b2 = *(const float4*)(br + 16);
            float4 b3 = *(const float4*)(br + 20);
            v16bf bfrag;
            pack8(bfrag, 0, b0, b1);
            pack8(bfrag, 8, b2, b3);
            acc[j] = __builtin_amdgcn_wmma_f32_16x16x32_bf16(
                         false, afrag, false, bfrag, (short)0, acc[j],
                         false, false);
        }
    }

    // C/D 16x16 f32 layout: lane = N, VGPR r = row (lanes 16-31 rows 8..15)
    const int mbase = m0 + (hf ? 8 : 0);
#pragma unroll
    for (int j = 0; j < NT; ++j) {
        const int n = (gn * NT + j) * 16 + lid;
        const float bv = (act != 0) ? bias[n] : 0.0f;
#pragma unroll
        for (int r = 0; r < 8; ++r) {
            float v = acc[j][r];
            if (act == 1) {                 // bias + softplus
                v += bv;
                v = (v > 20.0f) ? v : log1pf(__expf(v));
            } else if (act == 2) {          // bias only
                v += bv;
            }
            C[(size_t)(mbase + r) * ldc + n] = v;
        }
    }
}

// ---------------------------------------------------------------------------
// Zero-pad a rows x 16 slice (row stride srcld) into rows x 32 (cols 16..31 = 0)
// so the dt_proj GEMM runs the guard-free K=32 path.
// ---------------------------------------------------------------------------
__global__ __launch_bounds__(256)
void pad16to32_kernel(const float* __restrict__ src, int srcld,
                      float* __restrict__ dst, int rows)
{
    const int idx = blockIdx.x * blockDim.x + threadIdx.x;
    if (idx >= rows * 32) return;
    const int r = idx >> 5;
    const int k = idx & 31;
    dst[idx] = (k < 16) ? src[(size_t)r * srcld + k] : 0.0f;
}

// ---------------------------------------------------------------------------
// Residual add + RMSNorm over D=256.  One block per token row.
// ---------------------------------------------------------------------------
__global__ __launch_bounds__(DMODEL)
void rmsnorm_kernel(const float* __restrict__ x, int xstride,
                    const float* __restrict__ prev,   // nullable
                    const float* __restrict__ w,
                    float* __restrict__ res_out,
                    float* __restrict__ norm_out)
{
    __shared__ float sdata[DMODEL];
    const int row = blockIdx.x;
    const int c = threadIdx.x;
    float v = x[(size_t)row * xstride + c];
    if (prev) v += prev[(size_t)row * DMODEL + c];
    res_out[(size_t)row * DMODEL + c] = v;
    sdata[c] = v * v;
    __syncthreads();
    for (int s = DMODEL / 2; s > 0; s >>= 1) {
        if (c < s) sdata[c] += sdata[c + s];
        __syncthreads();
    }
    const float inv = rsqrtf(sdata[0] * (1.0f / DMODEL) + 1e-5f);
    norm_out[(size_t)row * DMODEL + c] = v * inv * w[c];
}

// ---------------------------------------------------------------------------
// Depthwise causal conv (K=4) + bias + SiLU.  xi = xz[:, 0:512].
// ---------------------------------------------------------------------------
__global__ __launch_bounds__(256)
void conv_silu_kernel(const float* __restrict__ xz,
                      const float* __restrict__ cw,
                      const float* __restrict__ cb,
                      float* __restrict__ xs)
{
    const int idx = blockIdx.x * blockDim.x + threadIdx.x;   // row*512 + c
    if (idx >= NTOK * DI) return;
    const int row = idx >> 9;
    const int c = idx & (DI - 1);
    const int t = row & (LSEQ - 1);                          // within batch
    float acc = cb[c];
#pragma unroll
    for (int j = 0; j < KC; ++j) {
        const int tt = t - (KC - 1) + j;
        if (tt >= 0)
            acc += cw[c * KC + j] * xz[(size_t)(row - (KC - 1) + j) * (2 * DI) + c];
    }
    xs[idx] = acc / (1.0f + __expf(-acc));                   // silu
}

// ---------------------------------------------------------------------------
// Fused selective scan.  Lane group of 16 owns one (b, channel); lane index
// s = state.  dA/dBu computed on the fly (never materialized: saves ~512 MB
// of traffic vs. the reference dataflow).  Contraction h·C via shfl_xor.
// Epilogue: + D*xs, * silu(z).
// ---------------------------------------------------------------------------
__global__ __launch_bounds__(256)
void scan_kernel(const float* __restrict__ dt,    // 2048 x 512 (softplus'd)
                 const float* __restrict__ xs,    // 2048 x 512
                 const float* __restrict__ dbc,   // 2048 x 48 (B @16, C @32)
                 const float* __restrict__ Alog,  // 512 x 16
                 const float* __restrict__ Dp,    // 512
                 const float* __restrict__ xz,    // 2048 x 1024 (z @ col 512)
                 float* __restrict__ y)           // 2048 x 512
{
    const int b = blockIdx.x;
    const int s = threadIdx.x & (DS - 1);
    const int c = blockIdx.y * (blockDim.x / DS) + (threadIdx.x >> 4);
    const float A = -__expf(Alog[c * DS + s]);
    const float D = Dp[c];
    float h = 0.0f;
    for (int t = 0; t < LSEQ; ++t) {
        const size_t row = (size_t)b * LSEQ + t;
        const float dtv = dt[row * DI + c];
        const float xsv = xs[row * DI + c];
        const float Bm = dbc[row * 48 + DR + s];
        const float Cm = dbc[row * 48 + DR + DS + s];
        h = __expf(dtv * A) * h + dtv * Bm * xsv;
        float p = h * Cm;
        p += __shfl_xor(p, 1);
        p += __shfl_xor(p, 2);
        p += __shfl_xor(p, 4);
        p += __shfl_xor(p, 8);
        if (s == 0) {
            const float z = xz[row * (2 * DI) + DI + c];
            y[row * DI + c] = (p + D * xsv) * (z / (1.0f + __expf(-z)));
        }
    }
}

// ---------------------------------------------------------------------------
// Final gating:  out[:, :256] = x1 ;  out[:, 256:] = (x2 + b)*sigmoid(s)*2
// ---------------------------------------------------------------------------
__global__ __launch_bounds__(256)
void final_gate_kernel(const float* __restrict__ x,
                       const float* __restrict__ sbuf,
                       const float* __restrict__ bbuf,
                       float* __restrict__ out)
{
    const int idx = blockIdx.x * blockDim.x + threadIdx.x;   // row*256 + col
    if (idx >= NTOK * DMODEL) return;
    const int row = idx >> 8;
    const int col = idx & (DMODEL - 1);
    const size_t base = (size_t)row * (2 * DMODEL);
    out[base + col] = x[base + col];
    const float sg = 1.0f / (1.0f + __expf(-sbuf[idx]));
    out[base + DMODEL + col] = (x[base + DMODEL + col] + bbuf[idx]) * sg * 2.0f;
}

// ---------------------------------------------------------------------------
// Host launcher
// ---------------------------------------------------------------------------
extern "C" void kernel_launch(void* const* d_in, const int* in_sizes, int n_in,
                              void* d_out, int out_size, void* d_ws, size_t ws_size,
                              hipStream_t stream)
{
    const float* x         = (const float*)d_in[0];   // (2,1024,512)
    const float* in_proj_w = (const float*)d_in[1];   // (4,1024,256)
    const float* conv_w    = (const float*)d_in[2];   // (4,512,4)
    const float* conv_b    = (const float*)d_in[3];   // (4,512)
    const float* x_proj_w  = (const float*)d_in[4];   // (4,48,512)
    const float* dt_proj_w = (const float*)d_in[5];   // (4,512,16)
    const float* dt_proj_b = (const float*)d_in[6];   // (4,512)
    const float* A_log     = (const float*)d_in[7];   // (4,512,16)
    const float* D_param   = (const float*)d_in[8];   // (4,512)
    const float* out_proj_w= (const float*)d_in[9];   // (4,256,512)
    const float* norm_w    = (const float*)d_in[10];  // (4,256)
    const float* norm_f_w  = (const float*)d_in[11];  // (256)
    const float* scale_w   = (const float*)d_in[12];  // (256,256)
    const float* scale_b   = (const float*)d_in[13];  // (256)
    const float* bias_w    = (const float*)d_in[14];  // (256,256)
    const float* bias_b    = (const float*)d_in[15];  // (256)
    float* out = (float*)d_out;

    // Workspace carve-up (floats).  Total ~32 MB -> lives in the 192 MB L2.
    float* ws = (float*)d_ws;
    float* res    = ws;                       // 2048*256
    float* hnorm  = res    + NTOK * DMODEL;   // 2048*256
    float* hidden = hnorm  + NTOK * DMODEL;   // 2048*256
    float* xz     = hidden + NTOK * DMODEL;   // 2048*1024
    float* xsb    = xz     + NTOK * 2 * DI;   // 2048*512
    float* dbc    = xsb    + NTOK * DI;       // 2048*48
    float* dtb    = dbc    + NTOK * 48;       // 2048*512
    float* yb     = dtb    + NTOK * DI;       // 2048*512
    float* sb     = yb     + NTOK * DI;       // 2048*256
    float* bb     = sb     + NTOK * DMODEL;   // 2048*256
    float* dtapad = bb     + NTOK * DMODEL;   // 2048*32  (dt activations, padded)
    float* dtwpad = dtapad + NTOK * 32;       // 512*32   (dt weights, padded)

    // (N/16) must be divisible by NT:  N=1024/512/256 -> NT=4,  N=48 -> NT=3.
    auto gemm = [&](const float* A, int lda, const float* W, int ldw,
                    float* C, int ldc, int M, int N, int K,
                    const float* bias, int act) {
        const int tilesN = N / 16;
        if (tilesN % 4 == 0) {
            const int jobs   = (M / 16) * (tilesN / 4);
            const int blocks = (jobs + 3) / 4;       // 4 waves / block
            gemm_wmma_kernel<4><<<blocks, 128, 0, stream>>>(
                A, lda, W, ldw, C, ldc, M, N, K, bias, act);
        } else {
            const int jobs   = (M / 16) * (tilesN / 3);
            const int blocks = (jobs + 3) / 4;
            gemm_wmma_kernel<3><<<blocks, 128, 0, stream>>>(
                A, lda, W, ldw, C, ldc, M, N, K, bias, act);
        }
    };

    for (int i = 0; i < NL; ++i) {
        const float* src    = (i == 0) ? x : hidden;
        const int    stride = (i == 0) ? 2 * DMODEL : DMODEL;
        const float* prev   = (i == 0) ? nullptr : res;

        // residual add + rmsnorm
        rmsnorm_kernel<<<NTOK, DMODEL, 0, stream>>>(src, stride, prev,
                                                    norm_w + i * DMODEL, res, hnorm);
        // in_proj: (2048x256)@(1024x256)^T -> xz (2048x1024)
        gemm(hnorm, DMODEL, in_proj_w + (size_t)i * 2 * DI * DMODEL, DMODEL,
             xz, 2 * DI, NTOK, 2 * DI, DMODEL, nullptr, 0);
        // depthwise conv + silu -> xs
        conv_silu_kernel<<<(NTOK * DI + 255) / 256, 256, 0, stream>>>(
            xz, conv_w + (size_t)i * DI * KC, conv_b + i * DI, xsb);
        // x_proj: (2048x512)@(48x512)^T -> dbc (2048x48)
        gemm(xsb, DI, x_proj_w + (size_t)i * 48 * DI, DI,
             dbc, 48, NTOK, 48, DI, nullptr, 0);
        // dt_proj + softplus: pad K 16 -> 32 on both operands, then WMMA GEMM
        pad16to32_kernel<<<(NTOK * 32 + 255) / 256, 256, 0, stream>>>(
            dbc, 48, dtapad, NTOK);
        pad16to32_kernel<<<(DI * 32 + 255) / 256, 256, 0, stream>>>(
            dt_proj_w + (size_t)i * DI * DR, DR, dtwpad, DI);
        gemm(dtapad, 32, dtwpad, 32, dtb, DI, NTOK, DI, 32,
             dt_proj_b + i * DI, 1);
        // fused selective scan (+ D*xs, * silu(z))
        scan_kernel<<<dim3(BQ, DI / 16), 256, 0, stream>>>(
            dtb, xsb, dbc, A_log + (size_t)i * DI * DS, D_param + i * DI, xz, yb);
        // out_proj: (2048x512)@(256x512)^T -> hidden (2048x256)
        gemm(yb, DI, out_proj_w + (size_t)i * DMODEL * DI, DI,
             hidden, DMODEL, NTOK, DMODEL, DI, nullptr, 0);
    }

    // final residual + rmsnorm (res updated in place: read+write same slot per thread)
    rmsnorm_kernel<<<NTOK, DMODEL, 0, stream>>>(hidden, DMODEL, res,
                                                norm_f_w, res, hnorm);
    // scale / bias projections
    gemm(hnorm, DMODEL, scale_w, DMODEL, sb, DMODEL,
         NTOK, DMODEL, DMODEL, scale_b, 2);
    gemm(hnorm, DMODEL, bias_w, DMODEL, bb, DMODEL,
         NTOK, DMODEL, DMODEL, bias_b, 2);
    // gate + concat
    final_gate_kernel<<<(NTOK * DMODEL + 255) / 256, 256, 0, stream>>>(x, sb, bb, out);
}
